// FusionAttention_66185446031635
// MI455X (gfx1250) — matrix operand order repaired and checked by reading, not compiled, not run on hardware
//
#include <hip/hip_runtime.h>
#include <hip/hip_bf16.h>

typedef __attribute__((ext_vector_type(16))) __bf16 v16bf;
typedef __attribute__((ext_vector_type(8)))  float  v8f;

#define WMMA_BF16(A_, B_, C_) \
  __builtin_amdgcn_wmma_f32_16x16x32_bf16(false, (A_), false, (B_), (short)0, (C_), false, false)

constexpr int Dm = 1024;   // model dim
constexpr int Sl = 512;    // sequence length
constexpr int NB = 16;     // batch
constexpr int NH = 16;     // heads
constexpr int HD = 64;     // head dim
// log2(10000)/32 for RoPE inv_freq = 10000^(-j/32) = 2^(-j*0.415241...)
#define ROPE_L2C 0.41524101186092030f

static __device__ __forceinline__ v8f zero8() {
  v8f z = {0.f,0.f,0.f,0.f,0.f,0.f,0.f,0.f};
  return z;
}

// ---------------------------------------------------------------------------
// Kernel 0: one-shot fp32 -> bf16 weight conversion (Wq,Wk,Wv,Wo).
// 16 MB read + 8 MB write (~1us @ 23.3 TB/s); removes 32 of 40 VALU cvts per
// 4-WMMA group in the GEMM inner loops.
// ---------------------------------------------------------------------------
__global__ __launch_bounds__(256)
void convert_weights(const float* __restrict__ w0, const float* __restrict__ w1,
                     const float* __restrict__ w2, const float* __restrict__ w3,
                     __bf16* __restrict__ o0, __bf16* __restrict__ o1,
                     __bf16* __restrict__ o2, __bf16* __restrict__ o3)
{
  const int z = blockIdx.z;
  const float* s = (z == 0) ? w0 : (z == 1) ? w1 : (z == 2) ? w2 : w3;
  __bf16* d      = (z == 0) ? o0 : (z == 1) ? o1 : (z == 2) ? o2 : o3;
  const int i = (blockIdx.x * 256 + threadIdx.x) * 8;
  #pragma unroll
  for (int k = 0; k < 8; ++k) d[i + k] = (__bf16)s[i + k];
}

// ---------------------------------------------------------------------------
// Kernel 1: y = x @ W^T + b, optional RoPE, store bf16 as [B,H,S,Hd].
// grid (N/64, M/128, 3{Q,K,V}); block (32,8): 8 waves, each wave = 16Mx64N.
// W is pre-converted bf16 (B fragments load with zero conversion cost).
// ---------------------------------------------------------------------------
__global__ __launch_bounds__(256)
void qkv_proj_rope(const float* __restrict__ xq, const float* __restrict__ xk,
                   const float* __restrict__ xv,
                   const __bf16* __restrict__ Wq, const float* __restrict__ bq,
                   const __bf16* __restrict__ Wk, const float* __restrict__ bk,
                   const __bf16* __restrict__ Wv, const float* __restrict__ bv,
                   __bf16* __restrict__ Qr, __bf16* __restrict__ Kr,
                   __bf16* __restrict__ Vr)
{
  const int z = blockIdx.z;
  const float*  x    = (z == 0) ? xq : (z == 1) ? xk : xv;
  const __bf16* W    = (z == 0) ? Wq : (z == 1) ? Wk : Wv;
  const float*  bias = (z == 0) ? bq : (z == 1) ? bk : bv;
  __bf16* out        = (z == 0) ? Qr : (z == 1) ? Kr : Vr;
  const bool rope = (z < 2);

  const int lane = threadIdx.x;
  const int c  = lane & 15;         // column-in-tile / A-row lane
  const int hl = lane >> 4;         // half-wave select
  const int mbase = blockIdx.y * 128 + threadIdx.y * 16;
  const int n0 = blockIdx.x * 64;   // head-aligned 64-wide N tile

  v8f acc[4];
  acc[0] = zero8(); acc[1] = zero8(); acc[2] = zero8(); acc[3] = zero8();

  const float* aprow = x + (size_t)(mbase + c) * Dm;   // A: lane c holds row M=c
  for (int kk = 0; kk < Dm; kk += 32) {
    // A fragment (16x32 bf16): elem i -> K = hl*16 + i
    v16bf a;
    const float* ap = aprow + kk + hl * 16;
    #pragma unroll
    for (int i = 0; i < 8; ++i) { a[i] = (__bf16)ap[i]; a[8 + i] = (__bf16)ap[8 + i]; }
    #pragma unroll
    for (int t = 0; t < 4; ++t) {
      // B fragment (32x16): lane n = c ; elem i -> K = hl*16 + i.  B[k][n] = W[n][k].
      const __bf16* bp = W + (size_t)(n0 + t * 16 + c) * Dm + kk + hl * 16;
      v16bf bfrag;
      #pragma unroll
      for (int i = 0; i < 16; ++i) bfrag[i] = bp[i];
      acc[t] = WMMA_BF16(a, bfrag, acc[t]);
    }
  }

  // bias epilogue (C layout: lane col N=c, elem v -> row M = v + hl*8)
  #pragma unroll
  for (int t = 0; t < 4; ++t) {
    const float bb = bias[n0 + t * 16 + c];
    #pragma unroll
    for (int v = 0; v < 8; ++v) acc[t][v] += bb;
  }

  if (rope) {
    // hd = t*16 + c; pair (hd, hd+32) = tiles (t, t+2), same lane/element.
    #pragma unroll
    for (int t = 0; t < 2; ++t) {
      const int j = t * 16 + c;                       // hd%32 (t in {0,1} -> hd<32)
      const float invf = exp2f(-(float)j * ROPE_L2C); // 10000^(-j/32)
      #pragma unroll
      for (int v = 0; v < 8; ++v) {
        const int row = mbase + v + hl * 8;
        const float pos = (float)(row & (Sl - 1));
        float sn, cs;
        __sincosf(pos * invf, &sn, &cs);
        const float x1 = acc[t][v], x2 = acc[t + 2][v];
        acc[t][v]     = x1 * cs - x2 * sn;            // hd < 32
        acc[t + 2][v] = x2 * cs + x1 * sn;            // hd >= 32
      }
    }
  }

  // store bf16 [B,H,S,Hd]
  #pragma unroll
  for (int t = 0; t < 4; ++t) {
    const int n_g = n0 + t * 16 + c;
    const int h = n_g >> 6, hd = n_g & 63;
    #pragma unroll
    for (int v = 0; v < 8; ++v) {
      const int row = mbase + v + hl * 8;
      const int b = row >> 9, s = row & (Sl - 1);
      out[((size_t)(b * NH + h) * Sl + s) * HD + hd] = (__bf16)acc[t][v];
    }
  }
}

// ---------------------------------------------------------------------------
// Kernel 2: flash attention per (b,h).  grid (B*H, S/64); block (32,4).
// Each wave owns 16 query columns; computes S^T = K.Q^T so the softmax (kv)
// axis lives in VGPR elements + one shfl_xor(16); then O^T = V^T.P.
// ---------------------------------------------------------------------------
__global__ __launch_bounds__(128)
void flash_attn(const __bf16* __restrict__ Qr, const __bf16* __restrict__ Kr,
                const __bf16* __restrict__ Vr, __bf16* __restrict__ Ao)
{
  __shared__ __bf16 Kt[32][72];   // 32 kv rows x 64 d (+8 pad)
  __shared__ __bf16 Vt[32][72];

  const int lane = threadIdx.x;
  const int wave = threadIdx.y;
  const int tid = wave * 32 + lane;
  const int c  = lane & 15;       // q column for this lane
  const int hl = lane >> 4;
  const int bh = blockIdx.x;
  const int q_g = blockIdx.y * 64 + wave * 16 + c;
  const size_t base = (size_t)bh * Sl * HD;
  const float scale = 0.125f;     // 1/sqrt(64)

  // Q B-fragments for S^T: lane n=q=c, elem i -> d = ks*32 + hl*16 + i
  v16bf qf[2];
  #pragma unroll
  for (int ks = 0; ks < 2; ++ks) {
    const __bf16* qp = Qr + base + (size_t)q_g * HD + ks * 32 + hl * 16;
    #pragma unroll
    for (int i = 0; i < 16; ++i) qf[ks][i] = qp[i];
  }

  v8f o[4];
  o[0] = zero8(); o[1] = zero8(); o[2] = zero8(); o[3] = zero8();
  float m_run = -3.0e38f, l_run = 0.f;

  for (int kb = 0; kb < Sl / 32; ++kb) {
    __syncthreads();
    // stage K,V tiles (32x64 bf16 each) into LDS as dword pairs
    for (int idx = tid; idx < 1024; idx += 128) {
      const int r = idx >> 5, c2 = idx & 31;
      const size_t g = base + (size_t)(kb * 32 + r) * HD + c2 * 2;
      *(unsigned int*)&Kt[r][c2 * 2] = *(const unsigned int*)(Kr + g);
      *(unsigned int*)&Vt[r][c2 * 2] = *(const unsigned int*)(Vr + g);
    }
    __syncthreads();
    if (kb + 1 < Sl / 32) {                    // gfx1250 global_prefetch_b8
      __builtin_prefetch(Kr + base + (size_t)((kb + 1) * 32) * HD + tid, 0, 1);
      __builtin_prefetch(Vr + base + (size_t)((kb + 1) * 32) * HD + tid, 0, 1);
    }

    // S^T tiles: M=kv (2 tiles of 16), N=q (16), K-dim=d (2 steps of 32)
    v8f st[2];
    st[0] = zero8(); st[1] = zero8();
    #pragma unroll
    for (int mt = 0; mt < 2; ++mt) {
      const int kvl = mt * 16 + c;             // A row M = kv
      #pragma unroll
      for (int ks = 0; ks < 2; ++ks) {
        v16bf af;
        const __bf16* kp = &Kt[kvl][ks * 32 + hl * 16];
        #pragma unroll
        for (int i = 0; i < 8; ++i) { af[i] = kp[i]; af[8 + i] = kp[8 + i]; }
        st[mt] = WMMA_BF16(af, qf[ks], st[mt]);
      }
    }
    #pragma unroll
    for (int mt = 0; mt < 2; ++mt)
      #pragma unroll
      for (int v = 0; v < 8; ++v) st[mt][v] *= scale;

    // online softmax over kv (elements + tiles + half-wave partner)
    float lmax = st[0][0];
    #pragma unroll
    for (int v = 0; v < 8; ++v) lmax = fmaxf(lmax, fmaxf(st[0][v], st[1][v]));
    lmax = fmaxf(lmax, __shfl_xor(lmax, 16, 32));
    const float m_new = fmaxf(m_run, lmax);
    const float corr = __expf(m_run - m_new);
    m_run = m_new;

    float p0[8], p1[8], lsum = 0.f;
    #pragma unroll
    for (int v = 0; v < 8; ++v) {
      p0[v] = __expf(st[0][v] - m_new);
      p1[v] = __expf(st[1][v] - m_new);
      lsum += p0[v] + p1[v];
    }
    lsum += __shfl_xor(lsum, 16, 32);
    l_run = l_run * corr + lsum;
    #pragma unroll
    for (int mt = 0; mt < 4; ++mt)
      #pragma unroll
      for (int v = 0; v < 8; ++v) o[mt][v] *= corr;

    // Assemble P B-fragment (K=kv 32, N=q): swap half-waves once per tile.
    float ot0[8], ot1[8];
    #pragma unroll
    for (int v = 0; v < 8; ++v) {
      ot0[v] = __shfl_xor(p0[v], 16, 32);
      ot1[v] = __shfl_xor(p1[v], 16, 32);
    }
    v16bf pb;
    #pragma unroll
    for (int i = 0; i < 8; ++i) {
      pb[i]     = (__bf16)(hl ? ot1[i] : p0[i]);   // K = hl*16 + i
      pb[8 + i] = (__bf16)(hl ? p1[i]  : ot0[i]);  // K = hl*16 + 8 + i
    }

    // O^T += V^T . P : A = V^T (M=hd tiles of 16, K=kv 32)
    #pragma unroll
    for (int mt = 0; mt < 4; ++mt) {
      const int hd = mt * 16 + c;
      v16bf vf;
      #pragma unroll
      for (int i = 0; i < 8; ++i) {
        vf[i]     = Vt[hl * 16 + i][hd];
        vf[8 + i] = Vt[hl * 16 + 8 + i][hd];
      }
      o[mt] = WMMA_BF16(vf, pb, o[mt]);
    }
  }

  const float inv = 1.0f / l_run;
  const int b = bh >> 4, h = bh & 15;
  #pragma unroll
  for (int mt = 0; mt < 4; ++mt)
    #pragma unroll
    for (int v = 0; v < 8; ++v) {
      const int hd = mt * 16 + v + hl * 8;     // O^T: lane N=q, elem -> M=hd
      Ao[(size_t)(b * Sl + q_g) * Dm + h * HD + hd] = (__bf16)(o[mt][v] * inv);
    }
}

// ---------------------------------------------------------------------------
// Kernel 3: y = attn_out(bf16) @ Wo^T(bf16) + bo -> fp32. grid (16,64); blk (32,8)
// ---------------------------------------------------------------------------
__global__ __launch_bounds__(256)
void out_proj(const __bf16* __restrict__ Ao, const __bf16* __restrict__ Wo,
              const float* __restrict__ bo, float* __restrict__ y)
{
  const int lane = threadIdx.x;
  const int c = lane & 15, hl = lane >> 4;
  const int mbase = blockIdx.y * 128 + threadIdx.y * 16;
  const int n0 = blockIdx.x * 64;

  v8f acc[4];
  acc[0] = zero8(); acc[1] = zero8(); acc[2] = zero8(); acc[3] = zero8();

  const __bf16* aprow = Ao + (size_t)(mbase + c) * Dm;
  for (int kk = 0; kk < Dm; kk += 32) {
    v16bf a;
    const __bf16* ap = aprow + kk + hl * 16;
    #pragma unroll
    for (int i = 0; i < 8; ++i) { a[i] = ap[i]; a[8 + i] = ap[8 + i]; }
    #pragma unroll
    for (int t = 0; t < 4; ++t) {
      const __bf16* bp = Wo + (size_t)(n0 + t * 16 + c) * Dm + kk + hl * 16;
      v16bf bfrag;
      #pragma unroll
      for (int i = 0; i < 16; ++i) bfrag[i] = bp[i];
      acc[t] = WMMA_BF16(a, bfrag, acc[t]);
    }
  }

  #pragma unroll
  for (int t = 0; t < 4; ++t) {
    const int n_g = n0 + t * 16 + c;
    const float bb = bo[n_g];
    #pragma unroll
    for (int v = 0; v < 8; ++v) {
      const int row = mbase + v + hl * 8;
      y[(size_t)row * Dm + n_g] = acc[t][v] + bb;
    }
  }
}

// ---------------------------------------------------------------------------
extern "C" void kernel_launch(void* const* d_in, const int* in_sizes, int n_in,
                              void* d_out, int out_size, void* d_ws, size_t ws_size,
                              hipStream_t stream)
{
  (void)in_sizes; (void)n_in; (void)out_size; (void)ws_size;
  const float* xq = (const float*)d_in[0];
  const float* xk = (const float*)d_in[1];
  const float* xv = (const float*)d_in[2];
  const float* Wq = (const float*)d_in[3];
  const float* bq = (const float*)d_in[4];
  const float* Wk = (const float*)d_in[5];
  const float* bk = (const float*)d_in[6];
  const float* Wv = (const float*)d_in[7];
  const float* bv = (const float*)d_in[8];
  const float* Wo = (const float*)d_in[9];
  const float* bo = (const float*)d_in[10];
  float* y = (float*)d_out;

  // workspace: Q,K,V [B,H,S,Hd] bf16 + attn_out [B,S,D] bf16 (4 x 16.78 MB)
  //          + 4 bf16 weight copies (4 x 2 MB) = ~75 MB
  const size_t qkv_elems = (size_t)NB * NH * Sl * HD;
  const size_t w_elems = (size_t)Dm * Dm;
  __bf16* Qr  = (__bf16*)d_ws;
  __bf16* Kr  = Qr + qkv_elems;
  __bf16* Vr  = Kr + qkv_elems;
  __bf16* Ao  = Vr + qkv_elems;
  __bf16* Wqb = Ao + qkv_elems;
  __bf16* Wkb = Wqb + w_elems;
  __bf16* Wvb = Wkb + w_elems;
  __bf16* Wob = Wvb + w_elems;

  convert_weights<<<dim3((unsigned)(w_elems / 2048), 1, 4), dim3(256), 0, stream>>>(
      Wq, Wk, Wv, Wo, Wqb, Wkb, Wvb, Wob);
  qkv_proj_rope<<<dim3(Dm / 64, (NB * Sl) / 128, 3), dim3(32, 8), 0, stream>>>(
      xq, xk, xv, Wqb, bq, Wkb, bk, Wvb, bv, Qr, Kr, Vr);
  flash_attn<<<dim3(NB * NH, Sl / 64), dim3(32, 4), 0, stream>>>(Qr, Kr, Vr, Ao);
  out_proj<<<dim3(Dm / 64, (NB * Sl) / 128), dim3(32, 8), 0, stream>>>(Ao, Wob, bo, y);
}
